// CrossAttentionFusion_41601053229206
// MI455X (gfx1250) — compile-verified, matrix operand order checked by reading, not run on hardware
//
#include <hip/hip_runtime.h>

// ---------------------------------------------------------------------------
// CrossAttentionFusion for MI455X (gfx1250): f16 WMMA GEMM pipeline
// V=3, B=8192, H=2048, NH=16, HD=128
// GEMM: 128x256 block tile, 8 waves x (64x64), BK=32, double-buffered LDS
// ---------------------------------------------------------------------------

typedef _Float16 h16;
typedef __attribute__((ext_vector_type(16))) _Float16 v16h;
typedef __attribute__((ext_vector_type(4)))  _Float16 v4h;
typedef __attribute__((ext_vector_type(8)))  float    v8f;
typedef __attribute__((ext_vector_type(4)))  unsigned int v4u;   // 16B move type (trivial, union-safe)

#define NV   3
#define NB   8192
#define NHID 2048
#define NHEAD 16
#define HDIM 128
#define NVH  (NV * NHID)   // 6144

#define BM 128
#define BN 256
#define BK 32
#define LDT 40             // padded LDS row stride in halves (80B) -> conflict-spread

struct GemmOffs {
    unsigned long long a[6];
    unsigned long long w[6];
    unsigned long long bi[6];
    unsigned long long c[6];
};

// ------------------------------- cast --------------------------------------
__global__ __launch_bounds__(256)
void cast_f32_f16(const float* __restrict__ src, h16* __restrict__ dst, long long n) {
    long long i = (long long)blockIdx.x * blockDim.x + threadIdx.x;
    long long stride = (long long)gridDim.x * blockDim.x;
    for (; i < n; i += stride) dst[i] = (h16)src[i];
}

// --------------------------- WMMA GEMM -------------------------------------
// C[M,N] = A[M,K](f16) * W[N,K](f16)^T + bias[N](f32)
// grid = (N/BN, M/BM, Z); per-z offsets in `offs`.
// Output: f16 to C16 (if non-null) and/or f32 to C32, row stride ldo.
__global__ __launch_bounds__(256)
void gemm_wmma_f16(const h16* __restrict__ Abase, const h16* __restrict__ Wbase,
                   const float* __restrict__ BiasBase,
                   h16* __restrict__ C16, float* __restrict__ C32,
                   int M, int N, int K, int ldo, GemmOffs offs)
{
    __shared__ h16 As[2][BM * LDT];   // 2 x 10240 B
    __shared__ h16 Bs[2][BN * LDT];   // 2 x 20480 B

    const int z = blockIdx.z;
    const h16*  A    = Abase   + offs.a[z];
    const h16*  W    = Wbase   + offs.w[z];
    const float* bias = BiasBase + offs.bi[z];

    const int m0 = blockIdx.y * BM;
    const int n0 = blockIdx.x * BN;

    const int tid  = threadIdx.x;
    const int wave = tid >> 5;        // 0..7
    const int lane = tid & 31;
    const int waveM = wave & 1;       // 2 waves along M -> 64 rows each
    const int waveN = wave >> 1;      // 4 waves along N -> 64 cols each

    // global->LDS staging mapping
    const int ldRowA = tid >> 1;        // 0..127, 2 threads/row, 32B each
    const int ldColA = (tid & 1) * 16;  // halves
    const int ldRowB = tid;             // 0..255, 1 thread/row, 64B each

    const h16* aRow = A + (long long)(m0 + ldRowA) * K + ldColA;
    const h16* bRow = W + (long long)(n0 + ldRowB) * K;

    v8f acc[4][4];
    {
        v8f zero = {};
        #pragma unroll
        for (int i = 0; i < 4; i++)
            #pragma unroll
            for (int j = 0; j < 4; j++) acc[i][j] = zero;
    }

    const int laneHalf = lane & 15;
    const int kbase    = (lane < 16) ? 0 : 8;   // 16-bit fragment K split

    union Frag { v16h v; v4u q[2]; };

    // ---- prologue: stage tile k0=0 into buffer 0 ----
    {
        v4u a0 = *(const v4u*)(aRow);
        v4u a1 = *(const v4u*)(aRow + 8);
        v4u b0 = *(const v4u*)(bRow);
        v4u b1 = *(const v4u*)(bRow + 8);
        v4u b2 = *(const v4u*)(bRow + 16);
        v4u b3 = *(const v4u*)(bRow + 24);
        *(v4u*)&As[0][ldRowA * LDT + ldColA]     = a0;
        *(v4u*)&As[0][ldRowA * LDT + ldColA + 8] = a1;
        *(v4u*)&Bs[0][ldRowB * LDT]      = b0;
        *(v4u*)&Bs[0][ldRowB * LDT + 8]  = b1;
        *(v4u*)&Bs[0][ldRowB * LDT + 16] = b2;
        *(v4u*)&Bs[0][ldRowB * LDT + 24] = b3;
    }
    __syncthreads();

    int buf = 0;
    for (int k0 = 0; k0 < K; k0 += BK) {
        const bool hasNext = (k0 + BK) < K;

        // stage next K-slab into registers while this slab computes
        v4u na0, na1, nb0, nb1, nb2, nb3;
        if (hasNext) {
            const h16* ag = aRow + (k0 + BK);
            const h16* bg = bRow + (k0 + BK);
            na0 = *(const v4u*)(ag);
            na1 = *(const v4u*)(ag + 8);
            nb0 = *(const v4u*)(bg);
            nb1 = *(const v4u*)(bg + 8);
            nb2 = *(const v4u*)(bg + 16);
            nb3 = *(const v4u*)(bg + 24);
            if (k0 + 2 * BK < K) {          // warm L2 for the slab after next
                __builtin_prefetch(ag + BK, 0, 3);
                __builtin_prefetch(bg + BK, 0, 3);
            }
        }

        // ---- compute from buf ----
        Frag fa[4], fb[4];
        #pragma unroll
        for (int mt = 0; mt < 4; mt++) {
            const int row = waveM * 64 + mt * 16 + laneHalf;
            fa[mt].q[0] = *(const v4u*)&As[buf][row * LDT + kbase];
            fa[mt].q[1] = *(const v4u*)&As[buf][row * LDT + kbase + 16];
        }
        #pragma unroll
        for (int nt = 0; nt < 4; nt++) {
            const int row = waveN * 64 + nt * 16 + laneHalf;
            fb[nt].q[0] = *(const v4u*)&Bs[buf][row * LDT + kbase];
            fb[nt].q[1] = *(const v4u*)&Bs[buf][row * LDT + kbase + 16];
        }
        #pragma unroll
        for (int mt = 0; mt < 4; mt++)
            #pragma unroll
            for (int nt = 0; nt < 4; nt++)
                acc[mt][nt] = __builtin_amdgcn_wmma_f32_16x16x32_f16(
                    false, fa[mt].v, false, fb[nt].v,
                    (short)0, acc[mt][nt], false, false);

        // ---- publish next slab into the other buffer ----
        if (hasNext) {
            const int nb = buf ^ 1;
            *(v4u*)&As[nb][ldRowA * LDT + ldColA]     = na0;
            *(v4u*)&As[nb][ldRowA * LDT + ldColA + 8] = na1;
            *(v4u*)&Bs[nb][ldRowB * LDT]      = nb0;
            *(v4u*)&Bs[nb][ldRowB * LDT + 8]  = nb1;
            *(v4u*)&Bs[nb][ldRowB * LDT + 16] = nb2;
            *(v4u*)&Bs[nb][ldRowB * LDT + 24] = nb3;
            __syncthreads();                 // single barrier per K-step
            buf = nb;
        }
    }

    // epilogue: C/D f32 16x16 layout -> lane%16 = col, (lane/16)*8 + r = row
    const int colBase = n0 + waveN * 64;
    const int rowBase = m0 + waveM * 64;
    #pragma unroll
    for (int mt = 0; mt < 4; mt++) {
        #pragma unroll
        for (int nt = 0; nt < 4; nt++) {
            const int col   = colBase + nt * 16 + (lane & 15);
            const float bb  = bias[col];
            const int mrow0 = rowBase + mt * 16 + (lane >> 4) * 8;
            if (C16) {
                h16* Cp = C16 + offs.c[z];
                #pragma unroll
                for (int r = 0; r < 8; r++)
                    Cp[(long long)(mrow0 + r) * ldo + col] = (h16)(acc[mt][nt][r] + bb);
            }
            if (C32) {
                float* Cp = C32 + offs.c[z];
                #pragma unroll
                for (int r = 0; r < 8; r++)
                    Cp[(long long)(mrow0 + r) * ldo + col] = acc[mt][nt][r] + bb;
            }
        }
    }
}

// ----------------------- attention (V-1 = 2 key views) ---------------------
// one wave32 per (view, batch-row, head); lane covers 4 of HD=128 elements
__global__ __launch_bounds__(256)
void attention_kernel(const h16* __restrict__ q2, const h16* __restrict__ k2,
                      const h16* __restrict__ v2, h16* __restrict__ ctx)
{
    const int lane = threadIdx.x & 31;
    const int wid  = blockIdx.x * 8 + (threadIdx.x >> 5);
    const int n  = wid % NHEAD;
    const int b  = (wid / NHEAD) % NB;
    const int vi = wid / (NHEAD * NB);

    const long long qoff = (((long long)vi * NB + b) * NHID) + n * HDIM + lane * 4;
    const v4h qv = *(const v4h*)(q2 + qoff);

    float sj[2];
    v4h   vv[2];
    #pragma unroll
    for (int j = 0; j < 2; j++) {
        const long long koff = ((((long long)vi * 2 + j) * NB + b) * NHID) + n * HDIM + lane * 4;
        const v4h kv = *(const v4h*)(k2 + koff);
        vv[j] = *(const v4h*)(v2 + koff);
        float p = 0.f;
        #pragma unroll
        for (int i = 0; i < 4; i++) p += (float)qv[i] * (float)kv[i];
        #pragma unroll
        for (int o = 16; o > 0; o >>= 1) p += __shfl_xor(p, o, 32);
        sj[j] = p * 0.08838834764831845f;   // 1/sqrt(HD)
    }
    const float m  = fmaxf(sj[0], sj[1]);
    const float e0 = __expf(sj[0] - m), e1 = __expf(sj[1] - m);
    const float inv = 1.f / (e0 + e1);
    const float w0 = e0 * inv, w1 = e1 * inv;

    v4h c;
    #pragma unroll
    for (int i = 0; i < 4; i++)
        c[i] = (h16)(w0 * (float)vv[0][i] + w1 * (float)vv[1][i]);
    *(v4h*)(ctx + qoff) = c;
}

// ----------------------- residual + LayerNorm ------------------------------
__global__ __launch_bounds__(256)
void residual_ln(const float* __restrict__ view0, const float* __restrict__ proj,
                 const float* __restrict__ gamma, const float* __restrict__ beta,
                 float* __restrict__ out)
{
    __shared__ float xs[NHID];
    __shared__ float red[2][8];
    const int b   = blockIdx.x;
    const int tid = threadIdx.x;

    float s = 0.f, ss = 0.f;
    for (int h = tid; h < NHID; h += 256) {
        const float x = view0[(long long)b * NHID + h] + proj[(long long)b * NHID + h];
        xs[h] = x; s += x; ss += x * x;
    }
    #pragma unroll
    for (int o = 16; o > 0; o >>= 1) { s += __shfl_xor(s, o, 32); ss += __shfl_xor(ss, o, 32); }
    const int wave = tid >> 5, lane = tid & 31;
    if (lane == 0) { red[0][wave] = s; red[1][wave] = ss; }
    __syncthreads();
    float ts = 0.f, tss = 0.f;
    #pragma unroll
    for (int i = 0; i < 8; i++) { ts += red[0][i]; tss += red[1][i]; }
    const float mu   = ts / NHID;
    const float var  = tss / NHID - mu * mu;
    const float rinv = rsqrtf(var + 1e-5f);
    for (int h = tid; h < NHID; h += 256)
        out[(long long)b * NHID + h] = (xs[h] - mu) * rinv * gamma[h] + beta[h];
}

// ---------------------------------------------------------------------------
extern "C" void kernel_launch(void* const* d_in, const int* in_sizes, int n_in,
                              void* d_out, int out_size, void* d_ws, size_t ws_size,
                              hipStream_t stream)
{
    const float* views = (const float*)d_in[0];
    const float* Wq  = (const float*)d_in[1];  const float* bq  = (const float*)d_in[2];
    const float* Wk  = (const float*)d_in[3];  const float* bk  = (const float*)d_in[4];
    const float* Wv  = (const float*)d_in[5];  const float* bv  = (const float*)d_in[6];
    const float* Wiq = (const float*)d_in[7];  const float* biq = (const float*)d_in[8];
    const float* Wik = (const float*)d_in[9];  const float* bik = (const float*)d_in[10];
    const float* Wiv = (const float*)d_in[11]; const float* biv = (const float*)d_in[12];
    const float* Wo  = (const float*)d_in[13]; const float* bo  = (const float*)d_in[14];
    const float* Wout= (const float*)d_in[15]; const float* bout= (const float*)d_in[16];
    const float* gamma = (const float*)d_in[17];
    const float* beta  = (const float*)d_in[18];
    float* out = (float*)d_out;

    char* ws = (char*)d_ws;
    size_t off = 0;
    auto a16 = [&](size_t elems) -> h16* {
        h16* p = (h16*)(ws + off);
        off += (elems * sizeof(h16) + 255) & ~(size_t)255;
        return p;
    };
    auto a32 = [&](size_t elems) -> float* {
        float* p = (float*)(ws + off);
        off += (elems * sizeof(float) + 255) & ~(size_t)255;
        return p;
    };

    const size_t VBH = (size_t)NV * NB * NHID;
    const size_t HH  = (size_t)NHID * NHID;
    const size_t VHH = (size_t)NV * HH;

    h16* hViews = a16(VBH);
    h16* hWq  = a16(VHH); h16* hWk  = a16(VHH); h16* hWv  = a16(VHH);
    h16* hWiq = a16(VHH); h16* hWik = a16(VHH); h16* hWiv = a16(VHH);
    h16* hWo  = a16(VHH); h16* hWout = a16((size_t)NHID * NVH);
    h16* hQ   = a16(VBH); h16* hK   = a16(VBH); h16* hVv  = a16(VBH); h16* hQ2 = a16(VBH);
    h16* hK2  = a16(VBH * 2); h16* hV2 = a16(VBH * 2);
    h16* hCtx = a16(VBH);
    h16* hFused = a16((size_t)NB * NVH);
    float* proj = a32((size_t)NB * NHID);

    auto cast = [&](const float* s, h16* d, size_t n) {
        size_t blocks = (n + 255) / 256; if (blocks > 8192) blocks = 8192;
        cast_f32_f16<<<dim3((unsigned)blocks), dim3(256), 0, stream>>>(s, d, (long long)n);
    };
    cast(views, hViews, VBH);
    cast(Wq, hWq, VHH);  cast(Wk, hWk, VHH);  cast(Wv, hWv, VHH);
    cast(Wiq, hWiq, VHH); cast(Wik, hWik, VHH); cast(Wiv, hWiv, VHH);
    cast(Wo, hWo, VHH);  cast(Wout, hWout, (size_t)NHID * NVH);

    const dim3 T(256);
    const dim3 GV(NHID / BN, NB / BM, NV);

    // per-view offsets: A=[V,B,H], W=[V,H,H], bias=[V,H], C=[V,B,H]
    GemmOffs ovh{};
    for (int z = 0; z < NV; z++) {
        ovh.a[z]  = (unsigned long long)z * NB * NHID;
        ovh.w[z]  = (unsigned long long)z * HH;
        ovh.bi[z] = (unsigned long long)z * NHID;
        ovh.c[z]  = (unsigned long long)z * NB * NHID;
    }

    // q / k / v projections, then q2
    gemm_wmma_f16<<<GV, T, 0, stream>>>(hViews, hWq, bq, hQ,  (float*)nullptr, NB, NHID, NHID, NHID, ovh);
    gemm_wmma_f16<<<GV, T, 0, stream>>>(hViews, hWk, bk, hK,  (float*)nullptr, NB, NHID, NHID, NHID, ovh);
    gemm_wmma_f16<<<GV, T, 0, stream>>>(hViews, hWv, bv, hVv, (float*)nullptr, NB, NHID, NHID, NHID, ovh);
    gemm_wmma_f16<<<GV, T, 0, stream>>>(hQ,     hWiq, biq, hQ2, (float*)nullptr, NB, NHID, NHID, NHID, ovh);

    // k2 / v2 : z -> (query view vi, slot j), source view o = others[vi][j]
    GemmOffs ok2{};
    for (int z = 0; z < NV * (NV - 1); z++) {
        const int vi = z / (NV - 1), j = z % (NV - 1);
        const int o2 = (j < vi) ? j : j + 1;
        ok2.a[z]  = (unsigned long long)o2 * NB * NHID;
        ok2.w[z]  = (unsigned long long)vi * HH;
        ok2.bi[z] = (unsigned long long)vi * NHID;
        ok2.c[z]  = (unsigned long long)z * NB * NHID;   // [V, V-1, B, H]
    }
    const dim3 G6(NHID / BN, NB / BM, NV * (NV - 1));
    gemm_wmma_f16<<<G6, T, 0, stream>>>(hK,  hWik, bik, hK2, (float*)nullptr, NB, NHID, NHID, NHID, ok2);
    gemm_wmma_f16<<<G6, T, 0, stream>>>(hVv, hWiv, biv, hV2, (float*)nullptr, NB, NHID, NHID, NHID, ok2);

    // softmax attention over the 2 key views
    const int totalWaves = NV * NB * NHEAD;
    attention_kernel<<<dim3(totalWaves / 8), T, 0, stream>>>(hQ2, hK2, hV2, hCtx);

    // out-proj, written straight into the fused [B, V*H] concat layout
    GemmOffs ofu{};
    for (int z = 0; z < NV; z++) {
        ofu.a[z]  = (unsigned long long)z * NB * NHID;
        ofu.w[z]  = (unsigned long long)z * HH;
        ofu.bi[z] = (unsigned long long)z * NHID;
        ofu.c[z]  = (unsigned long long)z * NHID;        // column block offset
    }
    gemm_wmma_f16<<<GV, T, 0, stream>>>(hCtx, hWo, bo, hFused, (float*)nullptr, NB, NHID, NHID, NVH, ofu);

    // fusion linear: [B, V*H] x Wout[H, V*H]^T -> f32 projected
    GemmOffs op{};
    gemm_wmma_f16<<<dim3(NHID / BN, NB / BM, 1), T, 0, stream>>>(
        hFused, hWout, bout, (h16*)nullptr, proj, NB, NHID, NVH, NHID, op);

    // residual with view 0 + LayerNorm
    residual_ln<<<dim3(NB), T, 0, stream>>>(views, proj, gamma, beta, out);
}
